// NeuralDecisionTree_1408749273376
// MI455X (gfx1250) — compile-verified, matrix operand order checked by reading, not run on hardware
//
#include <hip/hip_runtime.h>
#include <hip/hip_bf16.h>

#define BATCH 8192
#define D_IN 256
#define H1 128
#define H2 64
#define D_OUT 32
#define NLEAF 64
#define NNODE 63

typedef __attribute__((ext_vector_type(16))) __bf16 v16bf;
typedef __attribute__((ext_vector_type(8)))  __bf16 v8bf;
typedef __attribute__((ext_vector_type(8)))  float  v8f;

// ---- fragment loaders -------------------------------------------------------
// 16-bit A-matrix 16x32 layout (ISA 7.12.2): lane = half*16 + m, element e of
// v16bf maps to K = kk + (e<8 ? half*8+e : 16+half*8+(e-8)), row M = m.
// rowbase must already point at row m of a row-major bf16 matrix.
__device__ __forceinline__ v16bf load_a_frag(const __bf16* rowbase, int kk, int half) {
  union { v16bf v; v8bf h[2]; } u;
  u.h[0] = *(const v8bf*)(rowbase + kk + half * 8);
  u.h[1] = *(const v8bf*)(rowbase + kk + 16 + half * 8);
  return u.v;
}

// B fragments are pre-packed per 32x16 tile: 32 lanes * 16 bf16, contiguous.
__device__ __forceinline__ v16bf load_b_frag(const __bf16* blockbase, int lane) {
  union { v16bf v; v8bf h[2]; } u;
  const __bf16* p = blockbase + lane * 16;
  u.h[0] = *(const v8bf*)(p);
  u.h[1] = *(const v8bf*)(p + 8);
  return u.v;
}

#define WMMA_BF16(A, B, C) \
  __builtin_amdgcn_wmma_f32_16x16x32_bf16(false, (A), false, (B), (short)0, (C), false, false)

// ---- weight packer: W (L,K,N) fp32 row-major -> bf16 B-fragment blocks ------
// dst block order: ((l*nN + nt)*nK + ks) * 512, element order: lane*16 + e.
__global__ void pack_w_kernel(const float* __restrict__ W, __bf16* __restrict__ dst,
                              int L, int K, int N) {
  int idx = blockIdx.x * 256 + threadIdx.x;
  int nK = K >> 5, nN = N >> 4;
  int total = L * nN * nK * 512;
  if (idx >= total) return;
  int e    = idx & 15;
  int lane = (idx >> 4) & 31;
  int blk  = idx >> 9;
  int ks = blk % nK;  blk /= nK;
  int nt = blk % nN;  blk /= nN;
  int l  = blk;
  int half = lane >> 4, n = lane & 15;
  int k = ks * 32 + (e < 8 ? half * 8 + e : 16 + half * 8 + (e - 8));
  int col = nt * 16 + n;
  dst[idx] = (__bf16)W[((size_t)l * K + k) * N + col];
}

// ---- routing: fp32 sigmoid routes + leaf probabilities + x -> bf16 ----------
__global__ void route_kernel(const float* __restrict__ x, const float* __restrict__ Wr,
                             const float* __restrict__ br, float* __restrict__ p_out,
                             __bf16* __restrict__ x_bf) {
  __shared__ float rbuf[8][64];
  int wave = threadIdx.x >> 5, lane = threadIdx.x & 31;
  int b = blockIdx.x * 8 + wave;
  const float* xrow = x + (size_t)b * D_IN;

  // convert this x row to bf16 (8 elements per lane)
  for (int i = 0; i < 8; i++)
    x_bf[(size_t)b * D_IN + lane + i * 32] = (__bf16)xrow[lane + i * 32];

  // 63 routing nodes: lane handles node `lane` and node `lane+32` (if < 63)
  int n0 = lane, n1 = lane + 32;
  float a0 = br[n0];
  float a1 = (n1 < NNODE) ? br[n1] : 0.f;
  for (int kb = 0; kb < D_IN; kb += 32) {
    float xr = xrow[kb + lane];
    for (int j = 0; j < 32; j++) {
      float xv = __shfl(xr, j, 32);
      int k = kb + j;
      a0 += xv * Wr[k * NNODE + n0];
      if (n1 < NNODE) a1 += xv * Wr[k * NNODE + n1];
    }
  }
  rbuf[wave][n0] = 1.f / (1.f + __expf(-a0));
  if (n1 < NNODE) rbuf[wave][n1] = 1.f / (1.f + __expf(-a1));
  __syncthreads();

  // p[b][l] = prod over levels d of (bit ? 1-r : r), node = (2^d-1) + (l >> (6-d))
  for (int t = 0; t < 2; t++) {
    int l = lane + t * 32;
    float prob = 1.f;
    for (int d = 0; d < 6; d++) {
      int j = l >> (6 - d);
      int bit = (l >> (5 - d)) & 1;
      float r = rbuf[wave][(1 << d) - 1 + j];
      prob *= bit ? (1.f - r) : r;
    }
    p_out[(size_t)b * NLEAF + l] = prob;
  }
}

// ---- main kernel: 1 block per 16-row batch tile, 8 waves x 8 leaves each ----
__global__ __launch_bounds__(256) void tree_mlp_kernel(
    const __bf16* __restrict__ xbf, const float* __restrict__ p,
    const __bf16* __restrict__ W1p, const float* __restrict__ b1,
    const __bf16* __restrict__ W2p, const float* __restrict__ b2,
    const __bf16* __restrict__ W3p, const float* __restrict__ b3,
    float* __restrict__ out) {
  __shared__ __bf16 h1s[8][16][H1];   // 32 KB
  __shared__ __bf16 h2s[8][16][H2];   // 16 KB
  __shared__ float  red[8][16][D_OUT];// 16 KB

  int wave = threadIdx.x >> 5, lane = threadIdx.x & 31;
  int half = lane >> 4, m = lane & 15;
  int bt = blockIdx.x;

  // Load all 8 x A-fragments once; reused across all 8 leaves of this wave.
  const __bf16* xrow = xbf + (size_t)(bt * 16 + m) * D_IN;
  v16bf a1[8];
  for (int ks = 0; ks < 8; ks++) a1[ks] = load_a_frag(xrow, ks * 32, half);

  v8f oacc0 = {}, oacc1 = {};

  for (int li = 0; li < 8; li++) {
    int l = li * 8 + wave;

    // ---- GEMM1: x(16x256) @ W1[l](256x128) -> relu -> h1 (bf16 in LDS) ----
    const __bf16* w1l = W1p + (size_t)l * 8 * 8 * 512;
    for (int nt = 0; nt < 8; nt++) {
      v8f acc = {};
      for (int ks = 0; ks < 8; ks++) {
        v16bf bf = load_b_frag(w1l + (nt * 8 + ks) * 512, lane);
        acc = WMMA_BF16(a1[ks], bf, acc);
      }
      float bias = b1[l * H1 + nt * 16 + m];
      for (int r = 0; r < 8; r++) {
        float v = acc[r] + bias;
        v = v > 0.f ? v : 0.f;
        h1s[wave][r + 8 * half][nt * 16 + m] = (__bf16)v; // C layout: M=r+8*half, N=m
      }
    }

    // ---- GEMM2: h1(16x128) @ W2[l](128x64) -> relu -> h2 ----
    v16bf a2[4];
    for (int ks = 0; ks < 4; ks++) a2[ks] = load_a_frag(&h1s[wave][m][0], ks * 32, half);
    const __bf16* w2l = W2p + (size_t)l * 4 * 4 * 512;
    for (int nt = 0; nt < 4; nt++) {
      v8f acc = {};
      for (int ks = 0; ks < 4; ks++) {
        v16bf bf = load_b_frag(w2l + (nt * 4 + ks) * 512, lane);
        acc = WMMA_BF16(a2[ks], bf, acc);
      }
      float bias = b2[l * H2 + nt * 16 + m];
      for (int r = 0; r < 8; r++) {
        float v = acc[r] + bias;
        v = v > 0.f ? v : 0.f;
        h2s[wave][r + 8 * half][nt * 16 + m] = (__bf16)v;
      }
    }

    // ---- GEMM3: h2(16x64) @ W3[l](64x32), p-weighted accumulation ----
    v16bf a3[2];
    for (int ks = 0; ks < 2; ks++) a3[ks] = load_a_frag(&h2s[wave][m][0], ks * 32, half);
    const __bf16* w3l = W3p + (size_t)l * 2 * 2 * 512;
    float pv[8];
    for (int r = 0; r < 8; r++) pv[r] = p[(size_t)(bt * 16 + r + 8 * half) * NLEAF + l];
    for (int nt = 0; nt < 2; nt++) {
      v8f acc = {};
      for (int ks = 0; ks < 2; ks++) {
        v16bf bf = load_b_frag(w3l + (nt * 2 + ks) * 512, lane);
        acc = WMMA_BF16(a3[ks], bf, acc);
      }
      float bias = b3[l * D_OUT + nt * 16 + m];
      if (nt == 0) for (int r = 0; r < 8; r++) oacc0[r] += (acc[r] + bias) * pv[r];
      else         for (int r = 0; r < 8; r++) oacc1[r] += (acc[r] + bias) * pv[r];
    }
  }

  // ---- cross-wave (cross-leaf-group) reduction, then plain stores ----
  for (int r = 0; r < 8; r++) {
    red[wave][r + 8 * half][m]      = oacc0[r];
    red[wave][r + 8 * half][16 + m] = oacc1[r];
  }
  __syncthreads();
  for (int i = 0; i < 2; i++) {
    int e = threadIdx.x + i * 256;
    int row = e >> 5, col = e & 31;
    float s = 0.f;
    for (int w = 0; w < 8; w++) s += red[w][row][col];
    out[(size_t)(bt * 16 + row) * D_OUT + col] = s;
  }
}

// ---- launcher ---------------------------------------------------------------
extern "C" void kernel_launch(void* const* d_in, const int* in_sizes, int n_in,
                              void* d_out, int out_size, void* d_ws, size_t ws_size,
                              hipStream_t stream) {
  const float* x  = (const float*)d_in[0];
  const float* Wr = (const float*)d_in[1];
  const float* br = (const float*)d_in[2];
  const float* W1 = (const float*)d_in[3];
  const float* b1 = (const float*)d_in[4];
  const float* W2 = (const float*)d_in[5];
  const float* b2 = (const float*)d_in[6];
  const float* W3 = (const float*)d_in[7];
  const float* b3 = (const float*)d_in[8];
  float* out = (float*)d_out;

  char* ws = (char*)d_ws;
  size_t off = 0;
  __bf16* xbf = (__bf16*)(ws + off); off += (size_t)BATCH * D_IN * 2;        // 4 MB
  float*  pbf = (float*)(ws + off);  off += (size_t)BATCH * NLEAF * 4;       // 2 MB
  __bf16* W1p = (__bf16*)(ws + off); off += (size_t)NLEAF * D_IN * H1 * 2;   // 4 MB
  __bf16* W2p = (__bf16*)(ws + off); off += (size_t)NLEAF * H1 * H2 * 2;     // 1 MB
  __bf16* W3p = (__bf16*)(ws + off); off += (size_t)NLEAF * H2 * D_OUT * 2;  // 0.25 MB
  (void)ws_size; (void)in_sizes; (void)n_in; (void)out_size;

  int t1 = NLEAF * (H1 / 16) * (D_IN / 32) * 512;
  int t2 = NLEAF * (H2 / 16) * (H1 / 32) * 512;
  int t3 = NLEAF * (D_OUT / 16) * (H2 / 32) * 512;
  pack_w_kernel<<<(t1 + 255) / 256, 256, 0, stream>>>(W1, W1p, NLEAF, D_IN, H1);
  pack_w_kernel<<<(t2 + 255) / 256, 256, 0, stream>>>(W2, W2p, NLEAF, H1, H2);
  pack_w_kernel<<<(t3 + 255) / 256, 256, 0, stream>>>(W3, W3p, NLEAF, H2, D_OUT);
  route_kernel<<<BATCH / 8, 256, 0, stream>>>(x, Wr, br, pbf, xbf);
  tree_mlp_kernel<<<BATCH / 16, 256, 0, stream>>>(xbf, pbf, W1p, b1, W2p, b2, W3p, b3, out);
}
